// GridAttention_10282151707110
// MI455X (gfx1250) — matrix operand order repaired and checked by reading, not compiled, not run on hardware
//
#include <hip/hip_runtime.h>

typedef __attribute__((ext_vector_type(16))) _Float16     v16h;
typedef __attribute__((ext_vector_type(8)))  float        v8f;
typedef __attribute__((ext_vector_type(4)))  unsigned int v4u;
typedef __attribute__((ext_vector_type(8)))  int          v8i;
typedef __attribute__((ext_vector_type(4)))  int          v4i;

#define HW   16384
#define Hh   128
#define Wd   128
#define Lq   1536
#define LPAD 1544   // padded half-stride for bank-conflict-free b128 LDS reads

// ---------------------------------------------------------------------------
// K0: zero the per-batch sum/sumsq stats
// ---------------------------------------------------------------------------
__global__ void k_init(float* stats) {
    if (threadIdx.x < 4) stats[threadIdx.x] = 0.0f;
}

// ---------------------------------------------------------------------------
// K1: a = conv1x1(conv1x1(x, w1a, b1a), w1b, b1b); accumulate sum/sumsq per b
// ---------------------------------------------------------------------------
__global__ void __launch_bounds__(256) k_head(
    const float* __restrict__ x,  const float* __restrict__ w1a,
    const float* __restrict__ b1a,const float* __restrict__ w1b,
    const float* __restrict__ b1b,float* __restrict__ a,
    float* __restrict__ stats) {
    __shared__ float sw1a[256], sb1a[4], sw1b[16], sb1b[4];
    __shared__ float rsum[256], rsq[256];
    int t = threadIdx.x;
    sw1a[t] = w1a[t];
    if (t < 4)  { sb1a[t] = b1a[t]; sb1b[t] = b1b[t]; }
    if (t < 16) sw1b[t] = w1b[t];
    __syncthreads();

    int gid = blockIdx.x * 256 + t;          // 0 .. 32767 pixels
    int b   = gid >> 14;
    int p   = gid & (HW - 1);
    float tv0 = sb1a[0], tv1 = sb1a[1], tv2 = sb1a[2], tv3 = sb1a[3];
    const float* xb = x + (size_t)b * 64 * HW + p;
#pragma unroll 8
    for (int c = 0; c < 64; ++c) {
        if ((c & 15) == 0 && c + 16 < 64)
            __builtin_prefetch(xb + (c + 16) * HW, 0, 1);   // global_prefetch_b8
        float xv = xb[c * HW];
        tv0 += xv * sw1a[0 * 64 + c];
        tv1 += xv * sw1a[1 * 64 + c];
        tv2 += xv * sw1a[2 * 64 + c];
        tv3 += xv * sw1a[3 * 64 + c];
    }
    float lsum = 0.0f, lsq = 0.0f;
#pragma unroll
    for (int o = 0; o < 4; ++o) {
        float av = sb1b[o] + tv0 * sw1b[o*4+0] + tv1 * sw1b[o*4+1]
                           + tv2 * sw1b[o*4+2] + tv3 * sw1b[o*4+3];
        a[(size_t)(b * 4 + o) * HW + p] = av;
        lsum += av; lsq += av * av;
    }
    rsum[t] = lsum; rsq[t] = lsq;
    __syncthreads();
    for (int s = 128; s > 0; s >>= 1) {
        if (t < s) { rsum[t] += rsum[t + s]; rsq[t] += rsq[t + s]; }
        __syncthreads();
    }
    if (t == 0) {
        atomicAdd(&stats[b * 2 + 0], rsum[0]);
        atomicAdd(&stats[b * 2 + 1], rsq[0]);
    }
}

// ---------------------------------------------------------------------------
// K2: build normalized tokens (f16) and row-normalized qk (f16)
//     tok[bb,l,fw] = (apad[b,c,4gh+fh,4gw+fw]-mu[b])*rstd[b],
//     bb = b*16 + c*4 + fh,  l = gh*32+gw,  pad rows r<64 are zeros pre-norm
// ---------------------------------------------------------------------------
__global__ void __launch_bounds__(256) k_tok(
    const float* __restrict__ a, const float* __restrict__ wqk,
    const float* __restrict__ stats,
    _Float16* __restrict__ tokh, _Float16* __restrict__ qkh) {
    int gid = blockIdx.x * 256 + threadIdx.x;    // 0 .. 32*1536-1
    int bb  = gid / Lq;
    int l   = gid - bb * Lq;
    int b   = bb >> 4, c = (bb >> 2) & 3, fh = bb & 3;
    int gh  = l >> 5,  gw = l & 31;
    int r   = 4 * gh + fh;                       // padded row 0..191

    float sum = stats[b * 2], sq = stats[b * 2 + 1];
    const float invN = 1.0f / 98304.0f;          // 4*192*128 incl. zero pad
    float mu   = sum * invN;
    float rstd = rsqrtf(fmaxf(sq * invN - mu * mu, 0.0f) + 1e-5f);

    float tok[4];
#pragma unroll
    for (int fw = 0; fw < 4; ++fw) {
        float v = 0.0f;
        if (r >= 64)
            v = a[((size_t)(b * 4 + c) * Hh + (r - 64)) * Wd + 4 * gw + fw];
        tok[fw] = (v - mu) * rstd;
    }
    float qk[4];
#pragma unroll
    for (int j = 0; j < 4; ++j) {
        float s = 0.0f;
#pragma unroll
        for (int k = 0; k < 4; ++k) s += tok[k] * wqk[k * 4 + j];
        qk[j] = s;
    }
    float nrm = sqrtf(qk[0]*qk[0] + qk[1]*qk[1] + qk[2]*qk[2] + qk[3]*qk[3]);
    float inv = 1.0f / (nrm + 1e-8f);
    size_t base = (size_t)gid * 4;
#pragma unroll
    for (int k = 0; k < 4; ++k) {
        tokh[base + k] = (_Float16)tok[k];
        qkh [base + k] = (_Float16)(qk[k] * inv);
    }
}

// ---------------------------------------------------------------------------
// K3: attention. One workgroup (4 waves) per (bb, qtile<64).
//   tok tile (1536x4 f16 = 12288B) pulled into LDS by the Tensor Data Mover.
//   Phase A: S = Qn Qn^T via v_wmma_f32_16x16x32_f16 (K padded 4->32),
//            logits f16 -> LDS.
//   Phase B: row max / exp / row sum (unnormalized probs kept f16 in LDS).
//   Phase C: out = P @ tok via WMMA (K=32 chunks, N padded 4->16),
//            cross-wave reduce in LDS, scale by 1/rowsum, scatter to att.
// ---------------------------------------------------------------------------
__global__ void __launch_bounds__(128) k_attn(
    const _Float16* __restrict__ tokh, const _Float16* __restrict__ qkh,
    float* __restrict__ att) {
    __shared__ __align__(16) unsigned char smem[49408 + 12288 + 512 + 64 + 64];
    _Float16* lbuf   = (_Float16*)smem;                   // [16][LPAD] logits->probs
    _Float16* tokL   = (_Float16*)(smem + 49408);         // [1536][4] row-major
    float*    red    = (float*)(smem + 49408 + 12288);    // [16][8]
    float*    rowmax = red + 128;                         // [16]
    float*    rowinv = rowmax + 16;                       // [16]
    float*    accbuf = (float*)(smem + 49408);            // alias tokL (after barrier)

    int t    = threadIdx.x;
    int w    = t >> 5;          // wave 0..3
    int lane = t & 31;
    int g    = lane >> 4;       // half-wave
    int nl   = lane & 15;       // m for A / n for B,C

    int blk = blockIdx.x;       // 0..2047
    int bb  = blk >> 6;
    int qt  = blk & 63;         // only qtiles 0..63 survive the output crop
    int b = bb >> 4, c = (bb >> 2) & 3, fh = bb & 3;

    const _Float16* qkb = qkh  + (size_t)bb * Lq * 4;
    const _Float16* tkb = tokh + (size_t)bb * Lq * 4;

    // ---- TDM: async DMA tokh[bb] (1536x4 f16, contiguous 12288B) -> LDS.
    //      D# per CDNA5 ISA ch.8: 1-D tile of 6144 2-byte elements.
    if (w == 0) {
        unsigned long long gaddr = (unsigned long long)(const void*)tkb;
        unsigned           ldsa  = (unsigned)(size_t)(void*)tokL;
        v4u g0; v8i g1;
        v4i g2 = {0, 0, 0, 0}, g3 = {0, 0, 0, 0};
        g0[0] = 1u;                                  // count=1, user mode
        g0[1] = ldsa;                                // lds_addr
        g0[2] = (unsigned)gaddr;                     // global_addr[31:0]
        g0[3] = (unsigned)(gaddr >> 32) | (2u << 30);// global_addr[56:32] | type=2
        g1[0] = (1 << 16);                           // data_size=1 (2B), mask=0
        g1[1] = (int)(6144u << 16);                  // tensor_dim0[15:0]=6144
        g1[2] = (1 << 16);                           // tensor_dim1=1
        g1[3] = (int)(6144u << 16);                  // tile_dim0=6144
        g1[4] = 1;                                   // tile_dim1=1, tile_dim2=0
        g1[5] = 6144;                                // tensor_dim0_stride
        g1[6] = 0;
        g1[7] = 0;
#if __has_include(<hip/amd_detail/amd_gfx1250_TDM.h>)
        v8i g4 = {0, 0, 0, 0, 0, 0, 0, 0};           // clang-23 6-arg form
        __builtin_amdgcn_tensor_load_to_lds(g0, g1, g2, g3, g4, 0);
#else
        __builtin_amdgcn_tensor_load_to_lds(g0, g1, g2, g3, 0);
#endif
        __builtin_amdgcn_s_wait_tensorcnt(0);
    }

    // A fragment: Qn rows for this qtile (only K<4 nonzero -> g==0 half)
    v16h Aq = {};
    if (g == 0) {
        const _Float16* qr = qkb + (size_t)(qt * 16 + nl) * 4;
        Aq[0] = qr[0]; Aq[1] = qr[1]; Aq[2] = qr[2]; Aq[3] = qr[3];
    }
    const float rs = 0.17677669529663689f;   // 1/sqrt(32)

    // ---- Phase A
    for (int mt = w * 24; mt < w * 24 + 24; ++mt) {
        v16h Bq = {};
        if (g == 0) {
            const _Float16* mr = qkb + (size_t)(mt * 16 + nl) * 4;
            Bq[0] = mr[0]; Bq[1] = mr[1]; Bq[2] = mr[2]; Bq[3] = mr[3];
        }
        v8f Cz = {};
        v8f S = __builtin_amdgcn_wmma_f32_16x16x32_f16(
            false, Aq, false, Bq, (short)0, Cz, false, false);
        int colb = mt * 16 + nl;
#pragma unroll
        for (int r = 0; r < 8; ++r) {
            int M = r + 8 * g;
            lbuf[M * LPAD + colb] = (_Float16)((S[r] + 1.0f) * rs);
        }
    }
    __syncthreads();

    // ---- Phase B: softmax (store unnormalized p; keep 1/rowsum)
    {
        int row = t >> 3, ck = t & 7;
        int c0 = ck * 192;
        float mx = -1e30f;
        for (int j = 0; j < 192; ++j)
            mx = fmaxf(mx, (float)lbuf[row * LPAD + c0 + j]);
        red[row * 8 + ck] = mx;
        __syncthreads();
        if (t < 16) {
            float m2 = red[t * 8];
            for (int j = 1; j < 8; ++j) m2 = fmaxf(m2, red[t * 8 + j]);
            rowmax[t] = m2;
        }
        __syncthreads();
        float rm = rowmax[row];
        float s = 0.0f;
        for (int j = 0; j < 192; ++j) {
            float lv = (float)lbuf[row * LPAD + c0 + j];
            float p  = __expf(lv - rm);
            lbuf[row * LPAD + c0 + j] = (_Float16)p;
            s += p;
        }
        red[row * 8 + ck] = s;
        __syncthreads();
        if (t < 16) {
            float s2 = 0.0f;
            for (int j = 0; j < 8; ++j) s2 += red[t * 8 + j];
            rowinv[t] = 1.0f / s2;
        }
        __syncthreads();
    }

    // ---- Phase C: out(16x4) = P(16x1536) @ tok(1536x4), K in 32-chunks
    v8f acc = {};
    for (int ck2 = w * 12; ck2 < w * 12 + 12; ++ck2) {
        int kb = ck2 * 32;
        v16h Af, Bf;
#pragma unroll
        for (int i = 0; i < 16; ++i) {
            int kk = (i < 8) ? (8 * g + i) : (8 * g + i + 8);
            Af[i] = lbuf[nl * LPAD + kb + kk];
        }
#pragma unroll
        for (int i = 0; i < 16; ++i) {
            int kk = (i < 8) ? (8 * g + i) : (8 * g + i + 8);
            Bf[i] = (nl < 4) ? tokL[(kb + kk) * 4 + nl] : (_Float16)0.0f;
        }
        acc = __builtin_amdgcn_wmma_f32_16x16x32_f16(
            false, Af, false, Bf, (short)0, acc, false, false);
    }
    __syncthreads();                       // tokL reads done -> safe to alias
#pragma unroll
    for (int r = 0; r < 8; ++r) {
        int M = r + 8 * g;
        accbuf[w * 256 + M * 16 + nl] = acc[r];
    }
    __syncthreads();
    for (int e = t; e < 256; e += 128) {
        int M = e >> 4, N = e & 15;
        if (N < 4) {
            float v = accbuf[e] + accbuf[256 + e] + accbuf[512 + e] + accbuf[768 + e];
            v *= rowinv[M];
            int l = qt * 16 + M;
            int gh = l >> 5, gw = l & 31;   // 4*gh+c <= 127 guaranteed (qt<64)
            att[((size_t)(b * 4 + N) * Hh + (4 * gh + c)) * Wd + (4 * gw + fh)] = v;
        }
    }
}

// ---------------------------------------------------------------------------
// K4: y = x + wc @ (w2 @ att + b2) + bc.  WMMA for the 64x64 projection.
//     64 pixels per block (256 thr = 8 waves; 16 out tiles, 2 per wave).
// ---------------------------------------------------------------------------
__global__ void __launch_bounds__(256) k_proj(
    const float* __restrict__ att, const float* __restrict__ w2,
    const float* __restrict__ b2,  const float* __restrict__ wc,
    const float* __restrict__ bc,  const float* __restrict__ x,
    float* __restrict__ y) {
    __shared__ __align__(16) _Float16 zl[64 * 72];     // z[pixel][ch] f16
    __shared__ __align__(16) _Float16 wcTl[64 * 66];   // wc^T[k][n]   f16
    __shared__ float sb2[64], sbc[64], sw2[256];
    int t = threadIdx.x;
    if (t < 64) { sb2[t] = b2[t]; sbc[t] = bc[t]; }
    sw2[t] = w2[t];
    for (int idx = t; idx < 4096; idx += 256) {
        int n = idx >> 6, k = idx & 63;
        wcTl[k * 66 + n] = (_Float16)wc[n * 64 + k];
    }
    __syncthreads();

    int px0  = blockIdx.x * 64;
    int b    = px0 >> 14;
    int pimg = px0 & (HW - 1);
    for (int idx = t; idx < 4096; idx += 256) {
        int pl = idx >> 6, o = idx & 63;
        int p  = pimg + pl;
        float z = sb2[o];
#pragma unroll
        for (int k = 0; k < 4; ++k)
            z += sw2[o * 4 + k] * att[(size_t)(b * 4 + k) * HW + p];
        zl[pl * 72 + o] = (_Float16)z;
    }
    __syncthreads();

    int w = t >> 5, lane = t & 31, g = lane >> 4, nl = lane & 15;
#pragma unroll
    for (int ti = 0; ti < 2; ++ti) {
        int tt = w * 2 + ti;
        int Mt = tt >> 2, Nt = tt & 3;
        v8f acc = {};
#pragma unroll
        for (int kc = 0; kc < 64; kc += 32) {
            v16h Af, Bf;
#pragma unroll
            for (int i = 0; i < 16; ++i) {
                int kk = (i < 8) ? (8 * g + i) : (8 * g + i + 8);
                Af[i] = zl[(Mt * 16 + nl) * 72 + kc + kk];
            }
#pragma unroll
            for (int i = 0; i < 16; ++i) {
                int kk = (i < 8) ? (8 * g + i) : (8 * g + i + 8);
                Bf[i] = wcTl[(kc + kk) * 66 + Nt * 16 + nl];
            }
            acc = __builtin_amdgcn_wmma_f32_16x16x32_f16(
                false, Af, false, Bf, (short)0, acc, false, false);
        }
#pragma unroll
        for (int r = 0; r < 8; ++r) {
            int M  = r + 8 * g;
            int p  = pimg + Mt * 16 + M;
            int ch = Nt * 16 + nl;
            size_t gi = (size_t)(b * 64 + ch) * HW + p;
            y[gi] = acc[r] + sbc[ch] + x[gi];
        }
    }
}

// ---------------------------------------------------------------------------
// K5: h1 = relu(conv1x1(y, wf1, bf1))
// ---------------------------------------------------------------------------
__global__ void __launch_bounds__(256) k_ff1(
    const float* __restrict__ y, const float* __restrict__ wf1,
    const float* __restrict__ bf1, float* __restrict__ h1) {
    __shared__ float sw[1024], sb[16];
    int t = threadIdx.x;
    for (int i = t; i < 1024; i += 256) sw[i] = wf1[i];
    if (t < 16) sb[t] = bf1[t];
    __syncthreads();
    int gid = blockIdx.x * 256 + t;
    int b = gid >> 14, p = gid & (HW - 1);
    float h[16];
#pragma unroll
    for (int o = 0; o < 16; ++o) h[o] = sb[o];
    for (int c = 0; c < 64; ++c) {
        float v = y[(size_t)(b * 64 + c) * HW + p];
#pragma unroll
        for (int o = 0; o < 16; ++o) h[o] += v * sw[o * 64 + c];
    }
#pragma unroll
    for (int o = 0; o < 16; ++o)
        h1[(size_t)(b * 16 + o) * HW + p] = fmaxf(h[o], 0.0f);
}

// ---------------------------------------------------------------------------
// K6: dilated 3x3 (taps i-5,i-3,i-1 / j-2,j,j+2) + relu, then wf3 conv1x1,
//     final = y + ff  (in-place on d_out; each thread only touches its pixel)
// ---------------------------------------------------------------------------
__global__ void __launch_bounds__(256) k_ff2(
    const float* __restrict__ h1, const float* __restrict__ wf2,
    const float* __restrict__ bf2, const float* __restrict__ wf3,
    const float* __restrict__ bf3, float* __restrict__ out) {
    __shared__ float sw2[2304];            // [ci][kh][kw][o]
    __shared__ float sb2[16], sw3[1024], sb3[64];
    int t = threadIdx.x;
    for (int i = t; i < 2304; i += 256) {
        int o = i & 15, rest = i >> 4;     // rest = ci*9 + kh*3 + kw
        sw2[i] = wf2[o * 144 + rest];
    }
    if (t < 16) sb2[t] = bf2[t];
    for (int i = t; i < 1024; i += 256) sw3[i] = wf3[i];
    if (t < 64) sb3[t] = bf3[t];
    __syncthreads();

    int gid = blockIdx.x * 256 + t;
    int b = gid >> 14, p = gid & (HW - 1);
    int i = p >> 7, j = p & 127;
    float h2[16];
#pragma unroll
    for (int o = 0; o < 16; ++o) h2[o] = sb2[o];
    for (int ci = 0; ci < 16; ++ci) {
        const float* hb = h1 + (size_t)(b * 16 + ci) * HW;
#pragma unroll
        for (int kh = 0; kh < 3; ++kh) {
            int r = i - 5 + 2 * kh;
            if (r < 0) continue;           // r <= i-1 <= 126 always in range
#pragma unroll
            for (int kw = 0; kw < 3; ++kw) {
                int cc = j - 2 + 2 * kw;
                if (cc < 0 || cc > 127) continue;
                float v = hb[r * Wd + cc];
                const float* wr = &sw2[(ci * 9 + kh * 3 + kw) * 16];
#pragma unroll
                for (int o = 0; o < 16; ++o) h2[o] += v * wr[o];
            }
        }
    }
#pragma unroll
    for (int o = 0; o < 16; ++o) h2[o] = fmaxf(h2[o], 0.0f);
    for (int o = 0; o < 64; ++o) {
        float s = sb3[o];
#pragma unroll
        for (int c = 0; c < 16; ++c) s += sw3[o * 16 + c] * h2[c];
        size_t gi = (size_t)(b * 64 + o) * HW + p;
        out[gi] = out[gi] + s;
    }
}

// ---------------------------------------------------------------------------
extern "C" void kernel_launch(void* const* d_in, const int* in_sizes, int n_in,
                              void* d_out, int out_size, void* d_ws, size_t ws_size,
                              hipStream_t stream) {
    (void)in_sizes; (void)n_in; (void)out_size; (void)ws_size;
    const float* x   = (const float*)d_in[0];
    const float* w1a = (const float*)d_in[1];
    const float* b1a = (const float*)d_in[2];
    const float* w1b = (const float*)d_in[3];
    const float* b1b = (const float*)d_in[4];
    const float* wqk = (const float*)d_in[5];
    const float* w2  = (const float*)d_in[6];
    const float* b2  = (const float*)d_in[7];
    const float* wc  = (const float*)d_in[8];
    const float* bc  = (const float*)d_in[9];
    const float* wf1 = (const float*)d_in[10];
    const float* bf1 = (const float*)d_in[11];
    const float* wf2 = (const float*)d_in[12];
    const float* bf2 = (const float*)d_in[13];
    const float* wf3 = (const float*)d_in[14];
    const float* bf3 = (const float*)d_in[15];

    float* ws    = (float*)d_ws;
    float* stats = ws;                                   // 4 floats
    float* a     = ws + 64;                              // 131072
    _Float16* tokh = (_Float16*)(ws + 64 + 131072);      // 196608 halves
    _Float16* qkh  = (_Float16*)(ws + 64 + 131072 + 98304);
    float* att   = ws + 64 + 131072 + 2 * 98304;         // 131072
    float* h1    = att + 131072;                         // 524288
    float* y     = (float*)d_out;                        // y lives in d_out

    k_init<<<1, 64, 0, stream>>>(stats);
    k_head<<<128, 256, 0, stream>>>(x, w1a, b1a, w1b, b1b, a, stats);
    k_tok <<<192, 256, 0, stream>>>(a, wqk, stats, tokh, qkh);
    k_attn<<<2048, 128, 0, stream>>>(tokh, qkh, att);
    k_proj<<<512, 256, 0, stream>>>(att, w2, b2, wc, bc, x, y);
    k_ff1 <<<128, 256, 0, stream>>>(y, wf1, bf1, h1);
    k_ff2 <<<128, 256, 0, stream>>>(h1, wf2, bf2, wf3, bf3, y);
}
